// Reservoir_54116587930063
// MI455X (gfx1250) — matrix operand order repaired and checked by reading, not compiled, not run on hardware
//
#include <hip/hip_runtime.h>
#include <hip/hip_bf16.h>
#include <math.h>

// ---------------------------------------------------------------------------
// Echo-state network recurrence on MI455X (gfx1250, wave32, WMMA).
//   pre   = u_t @ W_in^T + s @ W^T + 0.01*n_t        (fused: K = 4096+64 = 4160)
//   s     = 0.7*s + 0.3*tanh(pre)                    (fp32 state)
//   out_t = s_t @ W_out^T + b                        (bf16 WMMA readout)
// W/W_in/W_out pre-swizzled once into the exact V_WMMA_F32_16X16X32_BF16
// B-operand lane layout; bf16 state history is both the recurrence ping-pong
// chain and the readout A-operand. Hot loops are software-pipelined so the
// compiler overlaps fragment loads with the WMMA accumulate chain.
// ---------------------------------------------------------------------------

typedef __attribute__((ext_vector_type(16))) __bf16 v16bf;
typedef __attribute__((ext_vector_type(8)))  __bf16 v8bf;
typedef __attribute__((ext_vector_type(8)))  float  v8f;

#define T_STEPS 512
#define BATCH   16
#define IN_DIM  64
#define RES     4096
#define OUT_DIM 64
#define KCAT    (RES + IN_DIM)   // 4160 fused K (reservoir + input)
#define NCHUNK  (KCAT / 32)      // 130 K-chunks of 32 for the recurrence
#define CMAIN   (RES / 32)       // 128 state K-chunks
#define KCH_OUT (RES / 32)       // 128 K-chunks for the readout
#define NTILES  (RES / 16)       // 256 N-tiles of the reservoir
#define LEAKF   0.3f
#define NOISEF  0.01f

// A-fragment for V_WMMA_*_16x16x32_bf16: lane holds M=lane&15, K-half h=lane>>4;
// elems 0..7 = K h*8+0..7, elems 8..15 = K 16+h*8+0..7. p points at the first
// group; the second is 16 elements (32 B) further along K.
__device__ __forceinline__ v16bf load_a_frag(const __bf16* p) {
    v8bf lo = *(const v8bf*)p;
    v8bf hi = *(const v8bf*)(p + 16);
    v16bf a;
#pragma unroll
    for (int e = 0; e < 8; ++e) { a[e] = lo[e]; a[e + 8] = hi[e]; }
    return a;
}

__device__ __forceinline__ float fast_tanh(float x) {
#if __has_builtin(__builtin_amdgcn_tanhf)
    return __builtin_amdgcn_tanhf(x);
#elif __has_builtin(__builtin_amdgcn_tanh_f32)
    return __builtin_amdgcn_tanh_f32(x);
#else
    return tanhf(x);
#endif
}

// ---------------------------------------------------------------------------
// Pre-swizzle W (R x R, row-major) and W_in (R x I) into B-fragment layout:
//   Wpk[(tile*NCHUNK + c)*512 + lane*16 + e] = Bsrc[k][n]
//   with n = tile*16 + (lane&15), k = c*32 + (lane>>4)*8 + e + (e&8)
//   Bsrc[k][n] = (k < R) ? W[n][k] : W_in[n][k-R]
// ---------------------------------------------------------------------------
__global__ __launch_bounds__(256) void pack_w_kernel(const float* __restrict__ W,
                                                     const float* __restrict__ Win,
                                                     __bf16* __restrict__ Wpk) {
    int gid = blockIdx.x * blockDim.x + threadIdx.x;
    if (gid >= NTILES * NCHUNK * 32) return;
    int lane  = gid & 31;
    int c     = (gid >> 5) % NCHUNK;
    int tile  = gid / (32 * NCHUNK);
    int n     = tile * 16 + (lane & 15);
    int kbase = c * 32 + (lane >> 4) * 8;
    __bf16* dst = Wpk + (size_t)gid * 16;
#pragma unroll
    for (int e = 0; e < 16; ++e) {
        int k = kbase + e + (e & 8);
        float v = (k < RES) ? W[(size_t)n * RES + k]
                            : Win[(size_t)n * IN_DIM + (k - RES)];
        dst[e] = (__bf16)v;
    }
}

// Same swizzle for W_out (O x R, row-major): 4 N-tiles x 128 K-chunks.
__global__ __launch_bounds__(256) void pack_wout_kernel(const float* __restrict__ Wout,
                                                        __bf16* __restrict__ WOpk) {
    int gid = blockIdx.x * blockDim.x + threadIdx.x;
    if (gid >= (OUT_DIM / 16) * KCH_OUT * 32) return;
    int lane  = gid & 31;
    int c     = (gid >> 5) % KCH_OUT;
    int tile  = gid / (32 * KCH_OUT);
    int o     = tile * 16 + (lane & 15);
    int kbase = c * 32 + (lane >> 4) * 8;
    __bf16* dst = WOpk + (size_t)gid * 16;
#pragma unroll
    for (int e = 0; e < 16; ++e)
        dst[e] = (__bf16)Wout[(size_t)o * RES + kbase + e + (e & 8)];
}

// Convert all u to bf16 (layout-preserving) and zero the t=0 history slot +
// the fp32 state. Runs every call => deterministic regardless of ws contents.
__global__ __launch_bounds__(256) void prep_misc_kernel(const float* __restrict__ u,
                                                        __bf16* __restrict__ u16,
                                                        __bf16* __restrict__ hist0,
                                                        float*  __restrict__ stF) {
    int gid = blockIdx.x * blockDim.x + threadIdx.x;
    if (gid < T_STEPS * BATCH * IN_DIM) u16[gid] = (__bf16)u[gid];
    if (gid < BATCH * RES) { hist0[gid] = (__bf16)0.0f; stF[gid] = 0.0f; }
}

// ---------------------------------------------------------------------------
// One recurrence step. Grid: 256 blocks (one 16x16 D-tile each, M=batch=16,
// N=16 reservoir cols). Block: 128 threads = 4 waves split-K.
// Main loop: exactly 32 state chunks per wave, software-pipelined so fragment
// loads for chunk i+1 are in flight while WMMA i executes. The 2 u_t chunks
// are a wave-uniform tail on waves 0/1 (no per-iteration pointer selects).
// ---------------------------------------------------------------------------
__global__ __launch_bounds__(128) void step_kernel(const __bf16* __restrict__ Wpk,
                                                   const __bf16* __restrict__ u16,
                                                   const float*  __restrict__ noise,
                                                   __bf16* __restrict__ hist,
                                                   float*  __restrict__ stF,
                                                   int t) {
    __shared__ float red[3][256];
    const int tid  = threadIdx.x;
    const int lane = tid & 31;
    const int wave = tid >> 5;
    const int tile = blockIdx.x;
    const int m    = lane & 15;   // A-row (batch) this lane feeds
    const int h    = lane >> 4;   // K-half selector of the A layout

    const __bf16* sA = hist + (size_t)t * BATCH * RES;      // s_t  [16][4096] bf16
    const __bf16* uA = u16  + (size_t)t * BATCH * IN_DIM;   // u_t  [16][64]   bf16
    const __bf16* bp = Wpk + (size_t)tile * NCHUNK * 512 + (size_t)lane * 16;

    // Per-wave chunk sequence: c = wave + 4*i, i = 0..31 (128/4 exact).
    const __bf16* bptr = bp + (size_t)wave * 512;            // +4*512 per iter
    const __bf16* aptr = sA + (size_t)m * RES + wave * 32 + h * 8; // +128 per iter

    v8f acc = {};
    v16bf bCur = *(const v16bf*)bptr;
    v16bf aCur = load_a_frag(aptr);
#pragma unroll 4
    for (int i = 0; i < 31; ++i) {
        bptr += 4 * 512;
        aptr += 4 * 32;
        v16bf bNxt = *(const v16bf*)bptr;       // in flight during the WMMA
        v16bf aNxt = load_a_frag(aptr);
        acc = __builtin_amdgcn_wmma_f32_16x16x32_bf16(false, aCur, false, bCur,
                                                      (short)0, acc, false, false);
        aCur = aNxt; bCur = bNxt;
    }
    acc = __builtin_amdgcn_wmma_f32_16x16x32_bf16(false, aCur, false, bCur,
                                                  (short)0, acc, false, false);

    // Input-projection tail: chunks 128+wave for waves 0 and 1 (wave-uniform).
    if (wave < 2) {
        v16bf aU = load_a_frag(uA + (size_t)m * IN_DIM + wave * 32 + h * 8);
        v16bf bU = *(const v16bf*)(bp + (size_t)(CMAIN + wave) * 512);
        acc = __builtin_amdgcn_wmma_f32_16x16x32_bf16(false, aU, false, bU,
                                                      (short)0, acc, false, false);
    }

    if (wave != 0) {
#pragma unroll
        for (int v = 0; v < 8; ++v) red[wave - 1][lane * 8 + v] = acc[v];
    }
    __syncthreads();
    if (wave == 0) {
        __bf16* histN = hist + (size_t)(t + 1) * BATCH * RES;
        const int r = tile * 16 + m;             // D layout: N = lane & 15
#pragma unroll
        for (int v = 0; v < 8; ++v) {
            float pre = acc[v] + red[0][lane * 8 + v] + red[1][lane * 8 + v]
                               + red[2][lane * 8 + v];
            int b = v + 8 * h;                   // D layout: M = vgpr + 8*(lane>>4)
            pre += NOISEF * noise[((size_t)t * BATCH + b) * RES + r];
            float old = stF[(size_t)b * RES + r];
            float ns  = (1.0f - LEAKF) * old + LEAKF * fast_tanh(pre);
            stF[(size_t)b * RES + r] = ns;
            histN[(size_t)b * RES + r] = (__bf16)ns;
        }
    }
}

// ---------------------------------------------------------------------------
// Readout: states[8192][4096] (bf16 history slots 1..512) @ W_out^T + b.
// Grid: 512 M-tiles x 4 N-tiles, one wave per block, 128 pipelined WMMAs.
// ---------------------------------------------------------------------------
__global__ __launch_bounds__(32) void readout_kernel(const __bf16* __restrict__ hist1,
                                                     const __bf16* __restrict__ WOpk,
                                                     const float*  __restrict__ bias,
                                                     float* __restrict__ out) {
    const int lane  = threadIdx.x;
    const int mtile = blockIdx.x;   // 0..511  (16 (t,b) rows each)
    const int ntile = blockIdx.y;   // 0..3    (16 output cols each)
    const int m = lane & 15;
    const int h = lane >> 4;

    const __bf16* aptr = hist1 + (size_t)(mtile * 16 + m) * RES + h * 8; // +32/chunk
    const __bf16* bptr = WOpk + (size_t)ntile * KCH_OUT * 512 + (size_t)lane * 16;

    v8f acc = {};
    v16bf bCur = *(const v16bf*)bptr;
    v16bf aCur = load_a_frag(aptr);
#pragma unroll 4
    for (int i = 0; i < KCH_OUT - 1; ++i) {
        bptr += 512;
        aptr += 32;
        v16bf bNxt = *(const v16bf*)bptr;
        v16bf aNxt = load_a_frag(aptr);
        acc = __builtin_amdgcn_wmma_f32_16x16x32_bf16(false, aCur, false, bCur,
                                                      (short)0, acc, false, false);
        aCur = aNxt; bCur = bNxt;
    }
    acc = __builtin_amdgcn_wmma_f32_16x16x32_bf16(false, aCur, false, bCur,
                                                  (short)0, acc, false, false);

#pragma unroll
    for (int v = 0; v < 8; ++v) {
        int row = mtile * 16 + v + 8 * h;        // (t*16 + b)
        int o   = ntile * 16 + m;
        out[(size_t)row * OUT_DIM + o] = acc[v] + bias[o];
    }
}

// ---------------------------------------------------------------------------
extern "C" void kernel_launch(void* const* d_in, const int* in_sizes, int n_in,
                              void* d_out, int out_size, void* d_ws, size_t ws_size,
                              hipStream_t stream) {
    const float* u     = (const float*)d_in[0];  // [512,16,64]
    const float* noise = (const float*)d_in[1];  // [512,16,4096]
    const float* Win   = (const float*)d_in[2];  // [4096,64]
    const float* W     = (const float*)d_in[3];  // [4096,4096]
    const float* Wout  = (const float*)d_in[4];  // [64,4096]
    const float* bias  = (const float*)d_in[5];  // [64]
    float* out = (float*)d_out;                  // [512,16,64]

    // Workspace carve-up (256 B aligned), ~103 MB total.
    char* ws = (char*)d_ws;
    size_t off = 0;
    auto carve = [&](size_t bytes) -> char* {
        char* p = ws + off;
        off += (bytes + 255) & ~(size_t)255;
        return p;
    };
    __bf16* Wpk  = (__bf16*)carve((size_t)NTILES * NCHUNK * 512 * 2);          // 34.1 MB
    __bf16* WOpk = (__bf16*)carve((size_t)(OUT_DIM / 16) * KCH_OUT * 512 * 2); // 0.5 MB
    __bf16* u16  = (__bf16*)carve((size_t)T_STEPS * BATCH * IN_DIM * 2);       // 1.0 MB
    __bf16* hist = (__bf16*)carve((size_t)(T_STEPS + 1) * BATCH * RES * 2);    // 67.2 MB
    float*  stF  = (float*) carve((size_t)BATCH * RES * 4);                    // 0.25 MB

    // --- one-time (per call) pre-swizzle + init ---
    {
        int threads = NTILES * NCHUNK * 32;                 // 1,064,960
        pack_w_kernel<<<(threads + 255) / 256, 256, 0, stream>>>(W, Win, Wpk);
    }
    {
        int threads = (OUT_DIM / 16) * KCH_OUT * 32;        // 16,384
        pack_wout_kernel<<<(threads + 255) / 256, 256, 0, stream>>>(Wout, WOpk);
    }
    {
        int threads = T_STEPS * BATCH * IN_DIM;             // 524,288 (>= BATCH*RES)
        prep_misc_kernel<<<(threads + 255) / 256, 256, 0, stream>>>(u, u16, hist, stF);
    }

    // --- sequential recurrence: 512 dependent launches (graph-captured) ---
    for (int t = 0; t < T_STEPS; ++t)
        step_kernel<<<NTILES, 128, 0, stream>>>(Wpk, u16, noise, hist, stF, t);

    // --- readout GEMM over the stored history (slots 1..512) ---
    readout_kernel<<<dim3(T_STEPS, OUT_DIM / 16), 32, 0, stream>>>(
        hist + (size_t)BATCH * RES, WOpk, bias, out);
}
